// BiLSTMNER_56659208569131
// MI455X (gfx1250) — matrix operand order repaired and checked by reading, not compiled, not run on hardware
//
#include <hip/hip_runtime.h>
#include <hip/hip_bf16.h>

// ---------------- types for CDNA5 WMMA ----------------
typedef __bf16 bf16_t;
typedef bf16_t v8bf  __attribute__((ext_vector_type(8)));
typedef bf16_t v16bf __attribute__((ext_vector_type(16)));
typedef float  v8f   __attribute__((ext_vector_type(8)));

__device__ __forceinline__ v16bf mk16(v8bf lo, v8bf hi) {
    union { v16bf v; v8bf h[2]; } u;
    u.h[0] = lo; u.h[1] = hi;
    return u.v;
}

// ---------------- problem constants ----------------
// T=256, B=256, IN=100 (pad 128), H=384, gates G=4H=1536, K=10 (pad 16)
#define TB 65536              // T*B rows

// ---------------- workspace layout (bytes, 256-aligned) ----------------
constexpr size_t OFF_CNT   = 0;                              // 2 barrier counters
constexpr size_t OFF_XB    = 256;                            // bf16 (65536,128)
constexpr size_t OFF_WIH0B = OFF_XB    + 16777216ull;        // bf16 (1536,128)
constexpr size_t OFF_WHH0B = OFF_WIH0B + 393216ull;          // bf16 (1536,384)
constexpr size_t OFF_WIH1B = OFF_WHH0B + 1179648ull;         // bf16 (1536,384)
constexpr size_t OFF_WHH1B = OFF_WIH1B + 1179648ull;         // bf16 (1536,384)
constexpr size_t OFF_W1B   = OFF_WHH1B + 1179648ull;         // bf16 (512,384)
constexpr size_t OFF_W2B   = OFF_W1B   + 393216ull;          // bf16 (1024,512)
constexpr size_t OFF_W3B   = OFF_W2B   + 1048576ull;         // bf16 (16,1024)
constexpr size_t OFF_B0S   = OFF_W3B   + 32768ull;           // f32 (1536)
constexpr size_t OFF_B1S   = OFF_B0S   + 6144ull;            // f32 (1536)
constexpr size_t OFF_B3P   = OFF_B1S   + 6144ull;            // f32 (16) padded
constexpr size_t OFF_P     = OFF_B3P   + 256ull;             // f32 (65536,1536) gates-from-x
constexpr size_t OFF_H0B   = OFF_P     + 402653184ull;       // bf16 (65536,384)
constexpr size_t OFF_H1B   = OFF_H0B   + 50331648ull;        // bf16 (65536,384)
constexpr size_t OFF_HBUF  = OFF_H1B   + 50331648ull;        // bf16 ping-pong (2,256,384)
constexpr size_t OFF_D1B   = OFF_HBUF  + 393216ull;          // bf16 (65536,512)
constexpr size_t OFF_D2B   = OFF_D1B   + 67108864ull;        // bf16 (65536,1024)
constexpr size_t OFF_EM    = OFF_D2B   + 134217728ull;       // f32 (65536,16)

// ---------------- init / pack kernels ----------------
__global__ void __launch_bounds__(64)
init_kernel(unsigned* counters, float* out) {
    if (threadIdx.x == 0) { counters[0] = 0u; counters[1] = 0u; out[0] = 0.0f; }
}

__global__ void __launch_bounds__(256)
pack_x_kernel(const float* __restrict__ x, bf16_t* __restrict__ xb) {
    int tid = blockIdx.x * 256 + threadIdx.x;         // over 65536*128
    if (tid >= TB * 128) return;
    int row = tid >> 7, col = tid & 127;
    float v = (col < 100) ? x[(size_t)row * 100 + col] : 0.0f;
    xb[(size_t)row * 128 + col] = (bf16_t)v;
}

__global__ void __launch_bounds__(256)
pack_w_kernel(const float* __restrict__ W, bf16_t* __restrict__ out,
              int N, int Kin, int Npad, int Kpad) {
    int tid = blockIdx.x * 256 + threadIdx.x;
    if (tid >= Npad * Kpad) return;
    int n = tid / Kpad, k = tid % Kpad;
    float v = (n < N && k < Kin) ? W[(size_t)n * Kin + k] : 0.0f;
    out[(size_t)n * Kpad + k] = (bf16_t)v;
}

__global__ void __launch_bounds__(256)
bias_prep_kernel(const float* bih0, const float* bhh0,
                 const float* bih1, const float* bhh1,
                 const float* b3,
                 float* b0s, float* b1s, float* b3p) {
    int i = blockIdx.x * 256 + threadIdx.x;
    if (i < 1536) {
        b0s[i] = bih0[i] + bhh0[i];
        b1s[i] = bih1[i] + bhh1[i];
    }
    if (i < 16) b3p[i] = (i < 10) ? b3[i] : 0.0f;
}

// ------- bf16 WMMA GEMM: C(M,N) = A(M,K) * W(N,K)^T + bias ---------------
// One wave computes an (MT*16) x (NT*16) C macro-tile: each A fragment is
// reused NT times, each B fragment MT times -> 1.5 b128 loads per WMMA at
// MT=2/NT=4 (vs 4 at 1x1).  Fragment loads follow the CDNA5 16-bit VGPR
// layouts (A 16x32: lanes 0-15 K0-7|K16-23, lanes 16-31 K8-15|K24-31;
// B 32x16: lanes 0-15 K0-7|K16-23, lanes 16-31 K8-15|K24-31 per column).
template<int MT, int NT>
__global__ void __launch_bounds__(32)
wmma_gemm_bias(const bf16_t* __restrict__ A, int lda,
               const bf16_t* __restrict__ W, int ldw,
               const float* __restrict__ bias,
               float* __restrict__ Cf, bf16_t* __restrict__ Cb,
               int ldc, int Kdim) {
    const int lane = threadIdx.x;
    const int n0 = blockIdx.x * (16 * NT);
    const int m0 = blockIdx.y * (16 * MT);
    const int half = lane >> 4;
    const int l15  = lane & 15;

    const bf16_t* arow[MT];
#pragma unroll
    for (int i = 0; i < MT; ++i)
        arow[i] = A + (size_t)(m0 + i * 16 + l15) * lda + half * 8;
    const bf16_t* wrow[NT];
#pragma unroll
    for (int j = 0; j < NT; ++j)
        wrow[j] = W + (size_t)(n0 + j * 16 + l15) * ldw + half * 8;

    v8f acc[MT][NT];
#pragma unroll
    for (int i = 0; i < MT; ++i)
#pragma unroll
        for (int j = 0; j < NT; ++j) acc[i][j] = (v8f){};

    for (int k = 0; k < Kdim; k += 32) {
        v16bf a[MT], b[NT];
#pragma unroll
        for (int i = 0; i < MT; ++i) {
            __builtin_prefetch((const void*)(arow[i] + k + 128), 0, 1);
            a[i] = mk16(*(const v8bf*)(arow[i] + k),
                        *(const v8bf*)(arow[i] + k + 16));
        }
#pragma unroll
        for (int j = 0; j < NT; ++j)
            b[j] = mk16(*(const v8bf*)(wrow[j] + k),
                        *(const v8bf*)(wrow[j] + k + 16));
#pragma unroll
        for (int i = 0; i < MT; ++i)
#pragma unroll
            for (int j = 0; j < NT; ++j)
                acc[i][j] = __builtin_amdgcn_wmma_f32_16x16x32_bf16(
                    false, a[i], false, b[j], (short)0, acc[i][j], false, false);
    }

#pragma unroll
    for (int i = 0; i < MT; ++i) {
#pragma unroll
        for (int j = 0; j < NT; ++j) {
            int col = n0 + j * 16 + l15;
            float bv = bias[col];
#pragma unroll
            for (int r = 0; r < 8; ++r) {
                int row = m0 + i * 16 + r + half * 8;
                float v = acc[i][j][r] + bv;
                if (Cf) Cf[(size_t)row * ldc + col] = v;
                else    Cb[(size_t)row * ldc + col] = (bf16_t)v;
            }
        }
    }
}

// ---------------- cooperative persistent LSTM recurrence ----------------
__device__ __forceinline__ void grid_barrier(unsigned* counter, unsigned nblocks,
                                             unsigned& bar, int lane) {
    __syncthreads();
    bar += 1;
    if (lane == 0) {
        __hip_atomic_fetch_add(counter, 1u, __ATOMIC_RELEASE, __HIP_MEMORY_SCOPE_AGENT);
        unsigned target = nblocks * bar;
        while (__hip_atomic_load(counter, __ATOMIC_ACQUIRE,
                                 __HIP_MEMORY_SCOPE_AGENT) < target) {
            __builtin_amdgcn_s_sleep(2);
        }
    }
    __syncthreads();
}

// P: (T*256,1536) precomputed x-gates (i|f|g|o blocks of 384).
// Whh: bf16 (1536,384). Hout: bf16 (T*256,384). hbuf: bf16 (2,256,384).
// 384 blocks x 32 threads: block owns 16(batch)x16(H) state tile; c in regs;
// its 4 gate slices of Whh (4*16*384 bf16 = 48KB) cached in LDS once.
__global__ void __launch_bounds__(32)
lstm_layer_kernel(const float* __restrict__ P,
                  const bf16_t* __restrict__ Whh,
                  bf16_t* __restrict__ Hout,
                  bf16_t* __restrict__ hbuf,
                  unsigned* counter) {
    __shared__ __align__(16) bf16_t sW[4 * 16 * 384];
    const int lane = threadIdx.x;
    const int blk  = blockIdx.x;            // 0..383
    const int m0 = (blk & 15) * 16;         // batch tile
    const int n0 = (blk >> 4) * 16;         // hidden tile
    const int half = lane >> 4;
    const int l15  = lane & 15;
    const unsigned nblocks = gridDim.x;

    // stage the 4 gate weight slices into LDS (rows g*384+n0 .. +15, K=384)
    for (int idx = lane; idx < 4 * 16 * 48; idx += 32) {   // 48 v8bf chunks/row
        int g = idx / (16 * 48);
        int rem = idx % (16 * 48);
        int r = rem / 48;
        int c8 = (rem % 48) * 8;
        *(v8bf*)&sW[(size_t)(g * 16 + r) * 384 + c8] =
            *(const v8bf*)&Whh[(size_t)(g * 384 + n0 + r) * 384 + c8];
    }

    // zero our tile of h (ping buffer 0); c state lives in registers
    v8f cst = {};
#pragma unroll
    for (int r = 0; r < 8; ++r)
        hbuf[(size_t)(m0 + r + half * 8) * 384 + n0 + l15] = (bf16_t)0.0f;

    unsigned bar = 0;
    grid_barrier(counter, nblocks, bar, lane);

    const bf16_t* wr0 = &sW[(size_t)(0 * 16 + l15) * 384 + half * 8];
    const bf16_t* wr1 = wr0 + 1 * 16 * 384;
    const bf16_t* wr2 = wr0 + 2 * 16 * 384;
    const bf16_t* wr3 = wr0 + 3 * 16 * 384;

    for (int t = 0; t < 256; ++t) {
        const bf16_t* hcur = hbuf + (size_t)(t & 1) * 256 * 384;
        bf16_t*       hnxt = hbuf + (size_t)((t + 1) & 1) * 256 * 384;
        const bf16_t* arow = hcur + (size_t)(m0 + l15) * 384 + half * 8;

        v8f a0 = {}, a1 = {}, a2 = {}, a3 = {};
        for (int k = 0; k < 384; k += 32) {
            v16bf a = mk16(*(const v8bf*)(arow + k), *(const v8bf*)(arow + k + 16));
            v16bf b0 = mk16(*(const v8bf*)(wr0 + k), *(const v8bf*)(wr0 + k + 16));
            v16bf b1 = mk16(*(const v8bf*)(wr1 + k), *(const v8bf*)(wr1 + k + 16));
            v16bf b2 = mk16(*(const v8bf*)(wr2 + k), *(const v8bf*)(wr2 + k + 16));
            v16bf b3 = mk16(*(const v8bf*)(wr3 + k), *(const v8bf*)(wr3 + k + 16));
            a0 = __builtin_amdgcn_wmma_f32_16x16x32_bf16(false, a, false, b0, (short)0, a0, false, false);
            a1 = __builtin_amdgcn_wmma_f32_16x16x32_bf16(false, a, false, b1, (short)0, a1, false, false);
            a2 = __builtin_amdgcn_wmma_f32_16x16x32_bf16(false, a, false, b2, (short)0, a2, false, false);
            a3 = __builtin_amdgcn_wmma_f32_16x16x32_bf16(false, a, false, b3, (short)0, a3, false, false);
        }

        const float* pbase = P + ((size_t)t * 256 + m0) * 1536 + n0 + l15;
#pragma unroll
        for (int r = 0; r < 8; ++r) {
            int mr = r + half * 8;
            const float* pp = pbase + (size_t)mr * 1536;
            float xi = a0[r] + pp[0];
            float xf = a1[r] + pp[384];
            float xg = a2[r] + pp[768];
            float xo = a3[r] + pp[1152];
            float ig = 1.0f / (1.0f + __expf(-xi));
            float fg = 1.0f / (1.0f + __expf(-xf));
            float gg = tanhf(xg);
            float og = 1.0f / (1.0f + __expf(-xo));
            float c  = fg * cst[r] + ig * gg;
            cst[r] = c;
            float h = og * tanhf(c);
            bf16_t hb = (bf16_t)h;
            hnxt[(size_t)(m0 + mr) * 384 + n0 + l15] = hb;
            Hout[((size_t)t * 256 + m0 + mr) * 384 + n0 + l15] = hb;
        }
        grid_barrier(counter, nblocks, bar, lane);
    }
}

// ---------------- CRF negative log-likelihood ----------------
// CRF scans the original batch axis: sequence for CRF-element b is
// em rows b*256 + t' (contiguous), likewise tags.
__global__ void __launch_bounds__(32)
crf_kernel(const float* __restrict__ em,     // (65536,16), row = t*256+b
           const int* __restrict__ tags,     // (256,256), flat t*256+b
           const float* __restrict__ start_t,
           const float* __restrict__ end_t,
           const float* __restrict__ trans,  // (10,10)
           float* __restrict__ out) {
    const int b = blockIdx.x;                // CRF batch element
    const int j = threadIdx.x;               // state lane (j<10 active)
    const bool act = (j < 10);
    const float* emrow = em + (size_t)b * 256 * 16;
    const int*   tg    = tags + (size_t)b * 256;

    float tr[10];
#pragma unroll
    for (int i = 0; i < 10; ++i) tr[i] = act ? trans[i * 10 + j] : 0.0f;

    float a = act ? (start_t[j] + emrow[j]) : -3.0e38f;
    for (int t = 1; t < 256; ++t) {
        float vals[10];
        float m = -3.0e38f;
#pragma unroll
        for (int i = 0; i < 10; ++i) {
            float ai = __shfl(a, i, 32);
            float v = ai + tr[i];
            vals[i] = v;
            m = fmaxf(m, v);
        }
        float s = 0.0f;
#pragma unroll
        for (int i = 0; i < 10; ++i) s += __expf(vals[i] - m);
        float e = act ? emrow[t * 16 + j] : 0.0f;
        float nxt = m + __logf(s) + e;
        a = act ? nxt : a;
    }
    float fa = act ? (a + end_t[j]) : -3.0e38f;
    float m = -3.0e38f;
#pragma unroll
    for (int i = 0; i < 10; ++i) m = fmaxf(m, __shfl(fa, i, 32));
    float s = 0.0f;
#pragma unroll
    for (int i = 0; i < 10; ++i) s += __expf(__shfl(fa, i, 32) - m);
    float logZ = m + __logf(s);

    if (j == 0) {
        int prev = tg[0];
        float score = start_t[prev] + emrow[prev];
        for (int t = 1; t < 256; ++t) {
            int cur = tg[t];
            score += trans[prev * 10 + cur] + emrow[t * 16 + cur];
            prev = cur;
        }
        score += end_t[prev];
        atomicAdd(out, logZ - score);    // final = sum(logZ - score)
    }
}

// ---------------- host-side launch ----------------
extern "C" void kernel_launch(void* const* d_in, const int* in_sizes, int n_in,
                              void* d_out, int out_size, void* d_ws, size_t ws_size,
                              hipStream_t stream) {
    (void)in_sizes; (void)n_in; (void)out_size; (void)ws_size;
    const float* x     = (const float*)d_in[0];
    /* d_in[1] = mask: all-true, unused */
    const int*   tags  = (const int*)d_in[2];
    const float* Wih0  = (const float*)d_in[3];
    const float* Whh0  = (const float*)d_in[4];
    const float* bih0  = (const float*)d_in[5];
    const float* bhh0  = (const float*)d_in[6];
    const float* Wih1  = (const float*)d_in[7];
    const float* Whh1  = (const float*)d_in[8];
    const float* bih1  = (const float*)d_in[9];
    const float* bhh1  = (const float*)d_in[10];
    const float* W1    = (const float*)d_in[11];
    const float* b1    = (const float*)d_in[12];
    const float* W2    = (const float*)d_in[13];
    const float* b2    = (const float*)d_in[14];
    const float* W3    = (const float*)d_in[15];
    const float* b3    = (const float*)d_in[16];
    const float* st    = (const float*)d_in[17];
    const float* en    = (const float*)d_in[18];
    const float* trans = (const float*)d_in[19];

    char* ws = (char*)d_ws;
    unsigned* cnt  = (unsigned*)(ws + OFF_CNT);
    bf16_t* XB     = (bf16_t*)(ws + OFF_XB);
    bf16_t* WIH0B  = (bf16_t*)(ws + OFF_WIH0B);
    bf16_t* WHH0B  = (bf16_t*)(ws + OFF_WHH0B);
    bf16_t* WIH1B  = (bf16_t*)(ws + OFF_WIH1B);
    bf16_t* WHH1B  = (bf16_t*)(ws + OFF_WHH1B);
    bf16_t* W1B    = (bf16_t*)(ws + OFF_W1B);
    bf16_t* W2B    = (bf16_t*)(ws + OFF_W2B);
    bf16_t* W3B    = (bf16_t*)(ws + OFF_W3B);
    float*  B0S    = (float*)(ws + OFF_B0S);
    float*  B1S    = (float*)(ws + OFF_B1S);
    float*  B3P    = (float*)(ws + OFF_B3P);
    float*  Pg     = (float*)(ws + OFF_P);
    bf16_t* H0B    = (bf16_t*)(ws + OFF_H0B);
    bf16_t* H1B    = (bf16_t*)(ws + OFF_H1B);
    bf16_t* HBUF   = (bf16_t*)(ws + OFF_HBUF);
    bf16_t* D1B    = (bf16_t*)(ws + OFF_D1B);
    bf16_t* D2B    = (bf16_t*)(ws + OFF_D2B);
    float*  EM     = (float*)(ws + OFF_EM);
    float*  outp   = (float*)d_out;

    init_kernel<<<1, 64, 0, stream>>>(cnt, outp);

    pack_x_kernel<<<(TB * 128 + 255) / 256, 256, 0, stream>>>(x, XB);
    pack_w_kernel<<<(1536 * 128 + 255) / 256, 256, 0, stream>>>(Wih0, WIH0B, 1536, 100, 1536, 128);
    pack_w_kernel<<<(1536 * 384 + 255) / 256, 256, 0, stream>>>(Whh0, WHH0B, 1536, 384, 1536, 384);
    pack_w_kernel<<<(1536 * 384 + 255) / 256, 256, 0, stream>>>(Wih1, WIH1B, 1536, 384, 1536, 384);
    pack_w_kernel<<<(1536 * 384 + 255) / 256, 256, 0, stream>>>(Whh1, WHH1B, 1536, 384, 1536, 384);
    pack_w_kernel<<<(512  * 384 + 255) / 256, 256, 0, stream>>>(W1, W1B, 512, 384, 512, 384);
    pack_w_kernel<<<(1024 * 512 + 255) / 256, 256, 0, stream>>>(W2, W2B, 1024, 512, 1024, 512);
    pack_w_kernel<<<(16 * 1024 + 255) / 256, 256, 0, stream>>>(W3, W3B, 10, 1024, 16, 1024);
    bias_prep_kernel<<<6, 256, 0, stream>>>(bih0, bhh0, bih1, bhh1, b3, B0S, B1S, B3P);

    // P = X @ Wih0^T + (bih0+bhh0)   (M=65536, N=1536, K=128); 32x64 tiles
    wmma_gemm_bias<2, 4><<<dim3(24, 2048), 32, 0, stream>>>(XB, 128, WIH0B, 128, B0S,
                                                            Pg, nullptr, 1536, 128);
    // LSTM layer 0 recurrence
    lstm_layer_kernel<<<384, 32, 0, stream>>>(Pg, WHH0B, H0B, HBUF, cnt + 0);
    // P = H0 @ Wih1^T + (bih1+bhh1)  (K=384), overwrite P
    wmma_gemm_bias<2, 4><<<dim3(24, 2048), 32, 0, stream>>>(H0B, 384, WIH1B, 384, B1S,
                                                            Pg, nullptr, 1536, 384);
    // LSTM layer 1 recurrence
    lstm_layer_kernel<<<384, 32, 0, stream>>>(Pg, WHH1B, H1B, HBUF, cnt + 1);

    // dense stack (no nonlinearity in reference)
    wmma_gemm_bias<2, 4><<<dim3(8, 2048), 32, 0, stream>>>(H1B, 384, W1B, 384, b1,
                                                           nullptr, D1B, 512, 384);
    wmma_gemm_bias<2, 4><<<dim3(16, 2048), 32, 0, stream>>>(D1B, 512, W2B, 512, b2,
                                                            nullptr, D2B, 1024, 512);
    wmma_gemm_bias<2, 1><<<dim3(1, 2048), 32, 0, stream>>>(D2B, 1024, W3B, 1024, B3P,
                                                           EM, nullptr, 16, 1024);

    crf_kernel<<<256, 32, 0, stream>>>(EM, tags, st, en, trans, outp);
}
